// SparseAttention_62612033241318
// MI455X (gfx1250) — matrix operand order repaired and checked by reading, not compiled, not run on hardware
//
#include <hip/hip_runtime.h>

typedef __attribute__((ext_vector_type(16))) _Float16 v16h;
typedef __attribute__((ext_vector_type(8)))  _Float16 v8h;
typedef __attribute__((ext_vector_type(8)))  float    v8f;

#define HEADS   8
#define NNODES  16384
#define DIM     256
#define NGLOB   4
#define TOTALN  (NNODES + NGLOB)        // 16388
#define MPAD    16400                   // 1025 * 16
#define ELOC    131072
#define EEXP    65536
#define NGEDGE  (NNODES * NGLOB)        // 65536
#define ETOT    (ELOC + NNODES + EEXP + NGEDGE + NGEDGE)  // 344064

// ---------------- workspace layout (bytes) ----------------
#define WS_XH      0u                       // MPAD*256 f16          = 8,396,800
#define WS_WQKVT   8396800u                 // 768*256 f16           =   393,216
#define WS_WOUTT   8790016u                 // 256*256 f16           =   131,072
#define WS_QKV     8921088u                 // MPAD*768 f32          = 50,380,800
#define WS_SCORES  59301888u                // ETOT*8 f32            = 11,010,048
#define WS_M       70311936u                // MPAD*8 f32            =    524,800
#define WS_DENOM   70836736u                // MPAD*8 f32            =    524,800
#define WS_ACC     71361536u                // MPAD*256 f32          = 16,793,600
#define WS_OH      88155136u                // NNODES*256 f16        =  8,388,608

// ---------------- init / convert kernels ----------------
__global__ void sa_init_kernel(float* __restrict__ acc, float* __restrict__ m,
                               float* __restrict__ denom) {
    int i = blockIdx.x * 256 + threadIdx.x;          // MPAD*256 threads
    acc[i] = 0.f;
    if (i < MPAD * 8) {
        denom[i] = 0.f;
        m[i] = -__builtin_inff();
    }
}

__global__ void sa_cvt_x_kernel(const float* __restrict__ x, const float* __restrict__ gt,
                                _Float16* __restrict__ xh) {
    int i = blockIdx.x * 256 + threadIdx.x;          // MPAD*256 threads
    int row = i >> 8, col = i & 255;
    float v = 0.f;
    if (row < NNODES)       v = x[i];
    else if (row < TOTALN)  v = gt[(row - NNODES) * 256 + col];
    xh[i] = (_Float16)v;
}

// W is (256, Ncols) row-major; wt is (Ncols, 256) row-major (N-major = B^T)
__global__ void sa_cvt_wT_kernel(const float* __restrict__ W, _Float16* __restrict__ wt,
                                 int Ncols) {
    int i = blockIdx.x * 256 + threadIdx.x;
    if (i >= Ncols * 256) return;
    int n = i >> 8, k = i & 255;
    wt[i] = (_Float16)W[k * Ncols + n];
}

// ---------------- WMMA GEMM: C(M x Ncols) = A(M x 256) * Bt^T + bias ----------------
// one wave computes a 16x64 tile (4 WMMA accumulators sharing one A fragment)
__global__ void sa_wmma_gemm_kernel(const _Float16* __restrict__ A,
                                    const _Float16* __restrict__ Bt,
                                    const float* __restrict__ bias,
                                    float* __restrict__ C,
                                    int mTiles, int nGroups, int ldc) {
    const int K = 256;
    int wave = threadIdx.x >> 5;
    int lane = threadIdx.x & 31;
    int w = blockIdx.x * (blockDim.x >> 5) + wave;
    if (w >= mTiles * nGroups) return;
    int mt = w / nGroups;
    int ng = w % nGroups;
    int row0 = mt * 16;
    int col0 = ng * 64;
    int l = lane & 15, hi = lane >> 4;

    v8f c0 = {}, c1 = {}, c2 = {}, c3 = {};
    const _Float16* arow = A + (size_t)(row0 + l) * K;
    const _Float16* bcol = Bt + (size_t)(col0 + l) * K;

    for (int kb = 0; kb < K; kb += 32) {
        // A fragment: lane(l,hi) holds M=l, K = kb + hi*16 .. +15 (contiguous)
        v8h alo = *(const v8h*)(arow + kb + hi * 16);
        v8h ahi = *(const v8h*)(arow + kb + hi * 16 + 8);
        v16h a;
#pragma unroll
        for (int i = 0; i < 8; ++i) { a[i] = alo[i]; a[8 + i] = ahi[i]; }

        // B fragment: lane(l,hi) holds N=col, K = kb + hi*16 .. +15 (contiguous in Bt)
        const _Float16* bb = bcol + kb + hi * 16;
        v16h b0 = *(const v16h*)(bb);
        v16h b1 = *(const v16h*)(bb + 16 * K);
        v16h b2 = *(const v16h*)(bb + 32 * K);
        v16h b3 = *(const v16h*)(bb + 48 * K);

        c0 = __builtin_amdgcn_wmma_f32_16x16x32_f16(false, a, false, b0, (short)0, c0, false, false);
        c1 = __builtin_amdgcn_wmma_f32_16x16x32_f16(false, a, false, b1, (short)0, c1, false, false);
        c2 = __builtin_amdgcn_wmma_f32_16x16x32_f16(false, a, false, b2, (short)0, c2, false, false);
        c3 = __builtin_amdgcn_wmma_f32_16x16x32_f16(false, a, false, b3, (short)0, c3, false, false);
    }

#pragma unroll
    for (int t = 0; t < 4; ++t) {
        v8f c = (t == 0) ? c0 : (t == 1) ? c1 : (t == 2) ? c2 : c3;
        int col = col0 + t * 16 + l;
        float bv = bias[col];
        float* cp = C + (size_t)(row0 + hi * 8) * ldc + col;
#pragma unroll
        for (int i = 0; i < 8; ++i) cp[(size_t)i * ldc] = c[i] + bv;
    }
}

// ---------------- edge helpers ----------------
__device__ __forceinline__ void sa_edge_sd(int e, const int* __restrict__ ei,
                                           const int* __restrict__ xe, int& s, int& d) {
    if (e < ELOC)                        { s = ei[e]; d = ei[ELOC + e]; }
    else if (e < ELOC + NNODES)          { s = d = e - ELOC; }
    else if (e < ELOC + NNODES + EEXP)   { int i = e - (ELOC + NNODES); s = xe[i]; d = xe[EEXP + i]; }
    else if (e < ELOC + NNODES + EEXP + NGEDGE) {
        int i = e - (ELOC + NNODES + EEXP); s = i >> 2; d = NNODES + (i & 3);
    } else {
        int i = e - (ELOC + NNODES + EEXP + NGEDGE); s = NNODES + (i & 3); d = i >> 2;
    }
}

__device__ __forceinline__ void sa_atomic_max_f32(float* addr, float val) {
    if (val >= 0.f) atomicMax((int*)addr, __float_as_int(val));
    else            atomicMin((unsigned int*)addr, __float_as_uint(val));
}

// ---------------- pass 1: scores + segment max ----------------
__global__ void sa_edge_score_kernel(const float* __restrict__ qkv,
                                     const int* __restrict__ ei, const int* __restrict__ xe,
                                     float* __restrict__ scores, float* __restrict__ m) {
    int wave = threadIdx.x >> 5, lane = threadIdx.x & 31;
    int e = blockIdx.x * 8 + wave;
    if (e >= ETOT) return;
    int s, d;
    sa_edge_sd(e, ei, xe, s, d);
    const float* Q  = qkv + (size_t)d * 768;        // Q block
    const float* Kp = qkv + (size_t)s * 768 + 256;  // K block
    const float scale = 0.17677669529663687f;       // 32^-0.5
#pragma unroll
    for (int h = 0; h < HEADS; ++h) {
        float p = Q[h * 32 + lane] * Kp[h * 32 + lane];
#pragma unroll
        for (int off = 16; off > 0; off >>= 1) p += __shfl_xor(p, off, 32);
        if (lane == 0) {
            float sc = p * scale;
            scores[(size_t)e * HEADS + h] = sc;
            sa_atomic_max_f32(&m[d * HEADS + h], sc);
        }
    }
}

// ---------------- pass 2: exp, denom, weighted V scatter ----------------
__global__ void sa_edge_accum_kernel(const float* __restrict__ qkv,
                                     const int* __restrict__ ei, const int* __restrict__ xe,
                                     const float* __restrict__ scores, const float* __restrict__ m,
                                     float* __restrict__ denom, float* __restrict__ acc) {
    int wave = threadIdx.x >> 5, lane = threadIdx.x & 31;
    int e = blockIdx.x * 8 + wave;
    if (e >= ETOT) return;
    int s, d;
    sa_edge_sd(e, ei, xe, s, d);
    float ev = 0.f;
    if (lane < HEADS) {
        ev = __expf(scores[(size_t)e * HEADS + lane] - m[d * HEADS + lane]);
        atomicAdd(&denom[d * HEADS + lane], ev);
    }
    const float* V = qkv + (size_t)s * 768 + 512;
    float* accd = acc + (size_t)d * 256;
#pragma unroll
    for (int h = 0; h < HEADS; ++h) {
        float evh = __shfl(ev, h, 32);
        atomicAdd(&accd[h * 32 + lane], evh * V[h * 32 + lane]);
    }
}

// ---------------- pass 3: normalize + f16 convert ----------------
__global__ void sa_normalize_kernel(const float* __restrict__ acc, const float* __restrict__ denom,
                                    _Float16* __restrict__ oh) {
    int i = blockIdx.x * 256 + threadIdx.x;          // NNODES*256 threads
    int n = i >> 8, c = i & 255;
    float v = acc[i] / denom[n * HEADS + (c >> 5)];
    oh[i] = (_Float16)v;
}

// ---------------- launch ----------------
extern "C" void kernel_launch(void* const* d_in, const int* in_sizes, int n_in,
                              void* d_out, int out_size, void* d_ws, size_t ws_size,
                              hipStream_t stream) {
    const float* x    = (const float*)d_in[0];
    const int*   ei   = (const int*)d_in[1];
    const int*   xe   = (const int*)d_in[2];
    const float* Wqkv = (const float*)d_in[3];
    const float* bqkv = (const float*)d_in[4];
    const float* Wout = (const float*)d_in[5];
    const float* bout = (const float*)d_in[6];
    const float* gt   = (const float*)d_in[7];
    (void)in_sizes; (void)n_in; (void)out_size; (void)ws_size;

    char* ws = (char*)d_ws;
    _Float16* xh     = (_Float16*)(ws + WS_XH);
    _Float16* wqT    = (_Float16*)(ws + WS_WQKVT);
    _Float16* woT    = (_Float16*)(ws + WS_WOUTT);
    float*    qkv    = (float*)(ws + WS_QKV);
    float*    scores = (float*)(ws + WS_SCORES);
    float*    m      = (float*)(ws + WS_M);
    float*    denom  = (float*)(ws + WS_DENOM);
    float*    acc    = (float*)(ws + WS_ACC);
    _Float16* oh     = (_Float16*)(ws + WS_OH);

    // init + f16 conversion
    sa_init_kernel<<<MPAD, 256, 0, stream>>>(acc, m, denom);
    sa_cvt_x_kernel<<<MPAD, 256, 0, stream>>>(x, gt, xh);
    sa_cvt_wT_kernel<<<768, 256, 0, stream>>>(Wqkv, wqT, 768);
    sa_cvt_wT_kernel<<<256, 256, 0, stream>>>(Wout, woT, 256);

    // QKV = x_full @ W_qkv + b_qkv   (1025 M-tiles x 12 N-groups = 12300 waves)
    sa_wmma_gemm_kernel<<<(1025 * 12 + 7) / 8, 256, 0, stream>>>(xh, wqT, bqkv, qkv, 1025, 12, 768);

    // edge softmax passes (wave per edge, 8 waves/block)
    sa_edge_score_kernel<<<ETOT / 8, 256, 0, stream>>>(qkv, ei, xe, scores, m);
    sa_edge_accum_kernel<<<ETOT / 8, 256, 0, stream>>>(qkv, ei, xe, scores, m, denom, acc);

    // normalize + convert
    sa_normalize_kernel<<<NNODES, 256, 0, stream>>>(acc, denom, oh);

    // out = attn_out @ W_out + b_out  (1024 x 4 groups = 4096 waves)
    sa_wmma_gemm_kernel<<<(1024 * 4) / 8, 256, 0, stream>>>(oh, woT, bout, (float*)d_out, 1024, 4, 256);
}
